// MLP_learner_17308718202969
// MI455X (gfx1250) — matrix operand order repaired and checked by reading, not compiled, not run on hardware
//
#include <hip/hip_runtime.h>
#include <stdint.h>

// ---------------------------------------------------------------------------
// MI455X (gfx1250) fused MLP -> cosine-kNN(33) graph builder.
// bf16 WMMA (v_wmma_f32_16x16x32_bf16) for the MLP GEMMs and the dominant
// 20000x20000x768 similarity GEMM (614 GFLOP). Xn (30 MB bf16) is L2-resident
// (192 MB L2); the fused sims+topk kernel streams it once per 64-row block
// (313 blocks -> ~9.4 GB L2 traffic), which bounds runtime, so the design
// maximizes rows/block under the 320 KB/WGP LDS budget. Top-k is lossless:
// sims tiles are dumped to a bank-conflict-free (pitch-65) LDS buffer and a
// lane-per-row merge of tile t-1 overlaps the WMMA compute of tile t.
// ---------------------------------------------------------------------------

#define NN    20000
#define DD    768
#define NTOP  33                 // k+1
#define NE    (NN * NTOP)        // 660000 edges (one direction)
#define LROW  (2 * NE)           // 1320000: edge_index row length

typedef __attribute__((ext_vector_type(16))) __bf16        bf16x16;
typedef __attribute__((ext_vector_type(8)))  float         f32x8;
typedef __attribute__((ext_vector_type(4)))  unsigned int  u32x4;
typedef __attribute__((ext_vector_type(4)))  float         f32x4;

union FragAB { bf16x16 bf; u32x4 u[2]; };

__device__ __forceinline__ unsigned short f32_to_bf16(float f) {
    union { float f; unsigned int u; } c; c.f = f;
    unsigned int u = c.u;
    u += 0x7FFFu + ((u >> 16) & 1u);           // round-to-nearest-even
    return (unsigned short)(u >> 16);
}

// ------------------------------- fp32 -> bf16 -------------------------------
__global__ __launch_bounds__(256) void cvt_f32_bf16(
    const float* __restrict__ in, unsigned short* __restrict__ out, int n)
{
    int i = blockIdx.x * blockDim.x + threadIdx.x;
    const int stride = gridDim.x * blockDim.x;
    for (; i < n; i += stride) out[i] = f32_to_bf16(in[i]);
}

// --------------------- GEMM: Out = act(A @ W^T + bias) ----------------------
// A [M,768] bf16 row-major, W [768,768] bf16 row-major (row n = weights of
// output feature n, i.e. B[k][n] = W[n][k]: B columns contiguous in K).
// One 16x16 output tile per wave; 24 x WMMA 16x16x32 bf16 over K.
template <bool RELU, bool OUTF32>
__global__ __launch_bounds__(256) void gemm_bias(
    const unsigned short* __restrict__ A,
    const unsigned short* __restrict__ W,
    const float* __restrict__ bias,
    void* __restrict__ Out, int M)
{
    const int lane = threadIdx.x & 31;
    const int half = lane >> 4;
    const int q    = lane & 15;
    const int tilesN = DD / 16;        // 48
    const int w = blockIdx.x * 8 + (threadIdx.x >> 5);
    const int tm = w / tilesN, tn = w % tilesN;
    if (tm * 16 >= M) return;
    const int m0 = tm * 16, n0 = tn * 16;

    f32x8 acc = {};
    const unsigned short* arow = A + (size_t)(m0 + q) * DD;   // A: lane = M
    const unsigned short* wrow = W + (size_t)(n0 + q) * DD;   // B: lane = N

#pragma unroll 4
    for (int kk = 0; kk < DD; kk += 32) {
        FragAB a, b;
        a.u[0] = *(const u32x4*)(arow + kk + half * 8);
        a.u[1] = *(const u32x4*)(arow + kk + 16 + half * 8);
        const unsigned short* bp = wrow + kk + half * 8;
        b.u[0] = *(const u32x4*)(bp);
        b.u[1] = *(const u32x4*)(bp + 16);
        acc = __builtin_amdgcn_wmma_f32_16x16x32_bf16(
                  false, a.bf, false, b.bf, (short)0, acc, false, false);
    }

    const float bn = bias[n0 + q];
#pragma unroll
    for (int e = 0; e < 8; ++e) {
        float v = acc[e] + bn;
        if (RELU) v = fmaxf(v, 0.f);
        const int m = m0 + half * 8 + e;       // C layout: VGPR e -> M = half*8+e
        if (OUTF32)
            ((float*)Out)[(size_t)m * DD + n0 + q] = v;
        else
            ((unsigned short*)Out)[(size_t)m * DD + n0 + q] = f32_to_bf16(v);
    }
}

// ------------------------- row L2-normalize -> bf16 -------------------------
__global__ __launch_bounds__(256) void rownorm_bf16(
    const float* __restrict__ E, unsigned short* __restrict__ Xn)
{
    const int lane = threadIdx.x & 31;
    const int row = blockIdx.x * 8 + (threadIdx.x >> 5);
    if (row >= NN) return;
    const float* er = E + (size_t)row * DD;
    float s = 0.f;
#pragma unroll
    for (int t = 0; t < 6; ++t) {
        f32x4 v = *(const f32x4*)(er + (lane + t * 32) * 4);
        s += v.x * v.x + v.y * v.y + v.z * v.z + v.w * v.w;
    }
#pragma unroll
    for (int m = 16; m >= 1; m >>= 1) s += __shfl_xor(s, m, 32);
    const float scale = 1.f / fmaxf(sqrtf(s), 1e-12f);
    unsigned int* outp = (unsigned int*)(Xn + (size_t)row * DD);
#pragma unroll
    for (int t = 0; t < 12; ++t) {
        const int p = lane + t * 32;
        const unsigned int pk = (unsigned int)f32_to_bf16(er[2 * p] * scale) |
                                ((unsigned int)f32_to_bf16(er[2 * p + 1] * scale) << 16);
        outp[p] = pk;
    }
}

// ------------------- fused sims (Xn @ Xn^T) + top-33 select -----------------
#define ROWS   64                     // query rows per block
#define CT     64                     // candidate cols per tile (4 col-waves)
#define NTILES ((NN + CT - 1) / CT)   // 313
#define APITCH 776                    // A_s row pitch (ushorts): 1552 B = 388
                                      // dwords == 4 mod 64 -> conflict-free
#define SPITCH 65                     // sims pitch (floats): (r*65+i)%64 distinct

__device__ __forceinline__ void merge_row(
    const float* __restrict__ Sm, int r, int base, int cmax,
    float* __restrict__ tV, int* __restrict__ tI, float& curMin, int& curPos)
{
    for (int i = 0; i < cmax; ++i) {
        const float v = Sm[r * SPITCH + i];
        if (v > curMin) {
            tV[r * NTOP + curPos] = v;
            tI[r * NTOP + curPos] = base + i;
            curMin = tV[r * NTOP]; curPos = 0;
            for (int j = 1; j < NTOP; ++j) {
                const float x = tV[r * NTOP + j];
                if (x < curMin) { curMin = x; curPos = j; }
            }
        }
    }
}

__global__ __launch_bounds__(256) void sims_topk(
    const unsigned short* __restrict__ Xn,
    int* __restrict__ eidx, float* __restrict__ ew)
{
    extern __shared__ char smem[];
    unsigned short* A_s = (unsigned short*)smem;           // 64*776*2 = 99328 B
    float* S0 = (float*)(smem + 99328);                    // 64*65*4  = 16640 B
    float* S1 = (float*)(smem + 99328 + 16640);            //            16640 B
    float* tV = (float*)(smem + 132608);                   // 64*33*4  =  8448 B
    int*   tI = (int*)  (smem + 141056);                   //             8448 B
                                                           // total 149504 B

    const int tid  = threadIdx.x;
    const int wave = tid >> 5, lane = tid & 31;
    const int half = lane >> 4, q = lane & 15;
    const int rw = wave >> 2;           // 0..1: row-wave (32 rows each)
    const int cg = wave & 3;            // 0..3: col-group (16 cols each)
    const int m0 = blockIdx.x * ROWS;
    const int validRows = min(ROWS, NN - m0);

    // Stage query rows into LDS with padded pitch.
    for (int idx = tid; idx < validRows * (DD / 8); idx += 256) {
        const int r = idx / (DD / 8), c = idx % (DD / 8);
        *(u32x4*)(A_s + r * APITCH + c * 8) =
            *(const u32x4*)(Xn + (size_t)(m0 + r) * DD + c * 8);
    }
    for (int i = tid; i < ROWS * NTOP; i += 256) { tV[i] = -3.0e38f; tI[i] = 0; }
    float curMin = -3.0e38f; int curPos = 0;    // live in merge lanes (tid<64)
    __syncthreads();

    const bool rowsOK = (rw * 32) < validRows;
    const unsigned short* a0p = A_s + (rw * 32 + q) * APITCH;
    const unsigned short* a1p = A_s + (rw * 32 + 16 + q) * APITCH;

    for (int t = 0; t < NTILES; ++t) {
        float* Sw = (t & 1) ? S1 : S0;
        const int c0w = t * CT + cg * 16;       // this wave's 16 candidate cols
        if (rowsOK && c0w < NN) {               // NN % 16 == 0: never partial
            if (c0w + CT < NN)
                __builtin_prefetch(Xn + (size_t)(c0w + CT + q) * DD, 0, 0);
            f32x8 acc0 = {}, acc1 = {};
            const unsigned short* brow = Xn + (size_t)(c0w + q) * DD + half * 8;
#pragma unroll 4
            for (int kk = 0; kk < DD; kk += 32) {
                FragAB a0, a1, b;
                b.u[0]  = *(const u32x4*)(brow + kk);
                b.u[1]  = *(const u32x4*)(brow + kk + 16);
                a0.u[0] = *(const u32x4*)(a0p + kk + half * 8);
                a0.u[1] = *(const u32x4*)(a0p + kk + 16 + half * 8);
                a1.u[0] = *(const u32x4*)(a1p + kk + half * 8);
                a1.u[1] = *(const u32x4*)(a1p + kk + 16 + half * 8);
                acc0 = __builtin_amdgcn_wmma_f32_16x16x32_bf16(
                           false, a0.bf, false, b.bf, (short)0, acc0, false, false);
                acc1 = __builtin_amdgcn_wmma_f32_16x16x32_bf16(
                           false, a1.bf, false, b.bf, (short)0, acc1, false, false);
            }
            // Dump tile to LDS (pitch 65 -> conflict-free strided stores).
            const int colL = cg * 16 + q;
#pragma unroll
            for (int e = 0; e < 8; ++e) {
                Sw[(rw * 32 + half * 8 + e) * SPITCH + colL]      = acc0[e];
                Sw[(rw * 32 + 16 + half * 8 + e) * SPITCH + colL] = acc1[e];
            }
        }
        // Lane-per-row merge of tile t-1 overlaps compute of tile t.
        if (t > 0 && tid < validRows) {
            const float* Sm = ((t - 1) & 1) ? S1 : S0;
            const int base = (t - 1) * CT;
            merge_row(Sm, tid, base, min(CT, NN - base), tV, tI, curMin, curPos);
        }
        __syncthreads();
    }

    if (tid < validRows) {
        // Merge the final tile (writes fenced by the loop's last barrier).
        {
            const float* Sm = ((NTILES - 1) & 1) ? S1 : S0;
            const int base = (NTILES - 1) * CT;
            merge_row(Sm, tid, base, min(CT, NN - base), tV, tI, curMin, curPos);
        }
        // Selection-sort 33 entries (descending) and emit edges:
        // edge_index ([2, 2*NE] int32: [rows|cols] ; [cols|rows]) + relu weights.
        const int r = tid, grow = m0 + r;
        for (int j = 0; j < NTOP; ++j) {
            float bm = -3.0e38f; int bp = 0;
            for (int i = 0; i < NTOP; ++i) {
                const float x = tV[r * NTOP + i];
                if (x > bm) { bm = x; bp = i; }
            }
            tV[r * NTOP + bp] = -3.0e38f;
            const int col = tI[r * NTOP + bp];
            const float wgt = fmaxf(bm, 0.f);
            const int e = grow * NTOP + j;
            eidx[e]             = grow;
            eidx[NE + e]        = col;
            eidx[LROW + e]      = col;
            eidx[LROW + NE + e] = grow;
            ew[e]      = wgt;
            ew[NE + e] = wgt;
        }
    }
}

// ---------------------------------------------------------------------------
extern "C" void kernel_launch(void* const* d_in, const int* in_sizes, int n_in,
                              void* d_out, int out_size, void* d_ws, size_t ws_size,
                              hipStream_t stream)
{
    const float* features = (const float*)d_in[0];
    const float* W1 = (const float*)d_in[1];
    const float* b1 = (const float*)d_in[2];
    const float* W2 = (const float*)d_in[3];
    const float* b2 = (const float*)d_in[4];

    char* ws = (char*)d_ws;                         // ~149 MB total
    unsigned short* Xb  = (unsigned short*)(ws);                 // 30.72 MB
    unsigned short* Hb  = (unsigned short*)(ws + 30720000);      // 30.72 MB
    unsigned short* Xn  = (unsigned short*)(ws + 61440000);      // 30.72 MB
    unsigned short* W1b = (unsigned short*)(ws + 92160000);      //  1.18 MB
    unsigned short* W2b = (unsigned short*)(ws + 93339648);      //  1.18 MB
    float*          E   = (float*)        (ws + 94519296);       // 61.44 MB

    cvt_f32_bf16<<<2048, 256, 0, stream>>>(features, Xb, NN * DD);
    cvt_f32_bf16<<<512, 256, 0, stream>>>(W1, W1b, DD * DD);
    cvt_f32_bf16<<<512, 256, 0, stream>>>(W2, W2b, DD * DD);

    // 20000/16 * 768/16 = 60000 wave-tiles, 8 waves/block -> 7500 blocks
    gemm_bias<true,  false><<<7500, 256, 0, stream>>>(Xb, W1b, b1, Hb, NN);
    gemm_bias<false, true ><<<7500, 256, 0, stream>>>(Hb, W2b, b2, E,  NN);

    rownorm_bf16<<<2500, 256, 0, stream>>>(E, Xn);

    const int smemBytes = 99328 + 16640 + 16640 + 8448 + 8448;   // 149504
    sims_topk<<<(NN + ROWS - 1) / ROWS, 256, smemBytes, stream>>>(
        Xn, (int*)d_out, (float*)d_out + 2 * LROW);
}
